// MultiheadGQA_7241314861605
// MI455X (gfx1250) — matrix operand order repaired and checked by reading, not compiled, not run on hardware
//
#include <hip/hip_runtime.h>
#include <hip/hip_bf16.h>

// ---------------------------------------------------------------------------
// GQA attention block for MI455X (gfx1250, wave32, WMMA).
// Pipeline: [cvt/transpose prep] -> [bf16 WMMA GEMM: Q,K,V] ->
//           [flash attention, WMMA QK^T + PV, online softmax] ->
//           [LayerNorm -> bf16] -> [bf16 WMMA GEMM: out proj -> fp32]
// ---------------------------------------------------------------------------

typedef __attribute__((ext_vector_type(16))) __bf16 v16bf;
typedef __attribute__((ext_vector_type(8)))  float  v8f;

#define D_MODEL   1024
#define SEQ_N     2048
#define BATCH     2
#define Q_HEADS   16
#define KV_HEADS  4
#define HEAD_DIM  64
#define KV_DIM    256
#define ROWS      (BATCH * SEQ_N)      // 4096
#define NEG_INF_L (-1e10f)

// ---------------------------------------------------------------------------
// Prep: fp32 -> bf16 copy (grid-stride)
// ---------------------------------------------------------------------------
__global__ __launch_bounds__(256)
void cvt_f32_to_bf16(const float* __restrict__ src, __bf16* __restrict__ dst, int n) {
    for (int i = blockIdx.x * blockDim.x + threadIdx.x; i < n; i += gridDim.x * blockDim.x)
        dst[i] = (__bf16)src[i];
}

// Prep: W[K,N] fp32 -> Wt[N,K] bf16 (so B-fragment K-pairs are contiguous)
__global__ __launch_bounds__(256)
void transpose_w_bf16(const float* __restrict__ W, __bf16* __restrict__ Wt, int K, int N) {
    int total = K * N;
    for (int i = blockIdx.x * blockDim.x + threadIdx.x; i < total; i += gridDim.x * blockDim.x) {
        int n = i / K;
        int k = i - n * K;
        Wt[i] = (__bf16)W[(size_t)k * N + n];
    }
}

// ---------------------------------------------------------------------------
// bf16 WMMA GEMM: C[M,N] = A[M,K] * Wt[N,K]^T + bias[N]
//   A row-major bf16, Wt row-major bf16 (i.e. original W transposed).
//   8 waves/block; wave w owns a 32x64 tile:
//     rows [blk.x*256 + w*32, +32), cols [blk.y*64, +64).
//   Per 32-K step: 2 A-fragments x 4 B-fragments = 8 back-to-back WMMAs
//   (B operands amortized across two M-subtiles -> ~2x arithmetic intensity).
//   store_bf16 != 0 : out bf16 with (val * out_scale); else fp32.
// Fragment layouts per CDNA5 ISA 7.12.2 (wave32):
//   A 16x32 bf16 : lane row = lane%16 ; VGPR v<4 holds K = k0+2v+8*half (pairs),
//                  VGPR 4+v holds K = k0+16+2v+8*half.
//   B 32x16 bf16 : lane col = lane%16 ; VGPR j holds K = k0+16*(j>>2)+8*half+2*(j&3) (pairs).
//   C 16x16 f32  : VGPR i -> row i + 8*half ; col = lane%16.
// ---------------------------------------------------------------------------
__global__ __launch_bounds__(256)
void gemm_bf16_wmma(const __bf16* __restrict__ A, const __bf16* __restrict__ Wt,
                    const float* __restrict__ bias, void* __restrict__ Cout,
                    int M, int N, int K, int store_bf16, float out_scale) {
    const int wave = threadIdx.x >> 5;
    const int lane = threadIdx.x & 31;
    const int half = lane >> 4;
    const int l16  = lane & 15;

    const int mtile = blockIdx.x * 256 + wave * 32;
    const int ntile = blockIdx.y * 64;

    v8f acc[2][4];
#pragma unroll
    for (int ms = 0; ms < 2; ++ms)
#pragma unroll
        for (int t = 0; t < 4; ++t) acc[ms][t] = v8f{};

    const __bf16* __restrict__ Arow0 = A + (size_t)(mtile +      l16) * K;
    const __bf16* __restrict__ Arow1 = A + (size_t)(mtile + 16 + l16) * K;
    const __bf16* __restrict__ Wr0   = Wt + (size_t)(ntile +      l16) * K;
    const __bf16* __restrict__ Wr1   = Wt + (size_t)(ntile + 16 + l16) * K;
    const __bf16* __restrict__ Wr2   = Wt + (size_t)(ntile + 32 + l16) * K;
    const __bf16* __restrict__ Wr3   = Wt + (size_t)(ntile + 48 + l16) * K;

    for (int k0 = 0; k0 < K; k0 += 32) {
        // hint next A cachelines into the hierarchy (global_prefetch_b8)
        __builtin_prefetch(Arow0 + k0 + 64, 0, 0);
        __builtin_prefetch(Arow1 + k0 + 64, 0, 0);

        v16bf a0, a1;
#pragma unroll
        for (int v = 0; v < 4; ++v) {
            const int kp = k0 + 2 * v + 8 * half;
            const int kq = k0 + 16 + 2 * v + 8 * half;
            a0[2 * v]     = Arow0[kp];
            a0[2 * v + 1] = Arow0[kp + 1];
            a0[2 * v + 8] = Arow0[kq];
            a0[2 * v + 9] = Arow0[kq + 1];
            a1[2 * v]     = Arow1[kp];
            a1[2 * v + 1] = Arow1[kp + 1];
            a1[2 * v + 8] = Arow1[kq];
            a1[2 * v + 9] = Arow1[kq + 1];
        }

        const __bf16* wr[4] = {Wr0, Wr1, Wr2, Wr3};
#pragma unroll
        for (int t = 0; t < 4; ++t) {
            v16bf b;
#pragma unroll
            for (int j = 0; j < 8; ++j) {
                const int kq = k0 + (j >> 2) * 16 + 8 * half + 2 * (j & 3);
                b[2 * j]     = wr[t][kq];
                b[2 * j + 1] = wr[t][kq + 1];
            }
            acc[0][t] = __builtin_amdgcn_wmma_f32_16x16x32_bf16(
                false, a0, false, b, (short)0, acc[0][t], false, false);
            acc[1][t] = __builtin_amdgcn_wmma_f32_16x16x32_bf16(
                false, a1, false, b, (short)0, acc[1][t], false, false);
        }
    }

    // epilogue: bias (+ optional scale), store
#pragma unroll
    for (int ms = 0; ms < 2; ++ms) {
#pragma unroll
        for (int t = 0; t < 4; ++t) {
            const int col = ntile + t * 16 + l16;
            const float bcol = bias[col];
#pragma unroll
            for (int i = 0; i < 8; ++i) {
                const int row = mtile + ms * 16 + i + 8 * half;
                float val = acc[ms][t][i] + bcol;
                if (store_bf16)
                    ((__bf16*)Cout)[(size_t)row * N + col] = (__bf16)(val * out_scale);
                else
                    ((float*)Cout)[(size_t)row * N + col] = val;
            }
        }
    }
}

// ---------------------------------------------------------------------------
// Flash attention (causal, GQA). One block = (batch bi, q-head H, 64-row qblock).
// 4 waves x 16 q-rows. Key blocks of 32, K/V staged in LDS (V transposed).
// Q is pre-scaled by 1/sqrt(64) at projection time.
// ---------------------------------------------------------------------------
__global__ __launch_bounds__(128)
void attn_flash_wmma(const __bf16* __restrict__ Q, const __bf16* __restrict__ Kb,
                     const __bf16* __restrict__ Vb, float* __restrict__ O) {
    const int qb   = blockIdx.x;           // 0..31 (n/64)
    const int head = blockIdx.y;           // 0..15
    const int bi   = blockIdx.z;           // 0..1
    const int kvh  = head & (KV_HEADS - 1);

    const int wave = threadIdx.x >> 5;     // 0..3
    const int lane = threadIdx.x & 31;
    const int half = lane >> 4;
    const int l16  = lane & 15;
    const int tid  = threadIdx.x;

    __shared__ __bf16 Ksh[32][72];         // [key][d]   (+pad)
    __shared__ __bf16 Vt[64][40];          // [d][key]   (+pad)
    __shared__ __bf16 Psh[4][16][32];      // per-wave P bounce (C->A transpose)

    const int qrow0 = qb * 64 + wave * 16;

    // load this wave's Q fragments once (A-layout, rows = lane%16)
    const __bf16* __restrict__ Qrow =
        Q + (size_t)(bi * SEQ_N + qrow0 + l16) * D_MODEL + head * HEAD_DIM;
    v16bf qa0, qa1;
#pragma unroll
    for (int v = 0; v < 4; ++v) {
        int kp = 2 * v + 8 * half;
        qa0[2 * v]     = Qrow[kp];
        qa0[2 * v + 1] = Qrow[kp + 1];
        qa0[2 * v + 8] = Qrow[16 + kp];
        qa0[2 * v + 9] = Qrow[16 + kp + 1];
        qa1[2 * v]     = Qrow[32 + kp];
        qa1[2 * v + 1] = Qrow[32 + kp + 1];
        qa1[2 * v + 8] = Qrow[48 + kp];
        qa1[2 * v + 9] = Qrow[48 + kp + 1];
    }

    float mrow[8], lrow[8];
#pragma unroll
    for (int i = 0; i < 8; ++i) { mrow[i] = -3.0e38f; lrow[i] = 0.0f; }
    v8f acc[4] = {v8f{}, v8f{}, v8f{}, v8f{}};

    const int nkb = 2 * (qb + 1);          // causal: keys <= qb*64+63
    for (int kb = 0; kb < nkb; ++kb) {
        const int k0 = kb * 32;

        // cooperative stage: K block [32 x 64] and V^T block [64 x 32]
        for (int idx = tid; idx < 32 * 64; idx += 128) {
            const int r = idx >> 6;        // key within block
            const int d = idx & 63;
            const size_t g = (size_t)(bi * SEQ_N + k0 + r) * KV_DIM + kvh * HEAD_DIM + d;
            Ksh[r][d] = Kb[g];
            Vt[d][r]  = Vb[g];
        }
        __syncthreads();

        // S = Q K^T for two 16-key tiles (K-dim = head_dim = 64 -> 2 wmma each)
        v8f s0 = v8f{}, s1 = v8f{};
#pragma unroll
        for (int dh = 0; dh < 2; ++dh) {
            v16bf b0, b1;
#pragma unroll
            for (int j = 0; j < 8; ++j) {
                int d = dh * 32 + (j >> 2) * 16 + 8 * half + 2 * (j & 3);
                b0[2 * j]     = Ksh[l16][d];
                b0[2 * j + 1] = Ksh[l16][d + 1];
                b1[2 * j]     = Ksh[16 + l16][d];
                b1[2 * j + 1] = Ksh[16 + l16][d + 1];
            }
            const v16bf qa = dh ? qa1 : qa0;
            s0 = __builtin_amdgcn_wmma_f32_16x16x32_bf16(false, qa, false, b0,
                                                         (short)0, s0, false, false);
            s1 = __builtin_amdgcn_wmma_f32_16x16x32_bf16(false, qa, false, b1,
                                                         (short)0, s1, false, false);
        }

        // causal mask
        const int key0 = k0 + l16;
        const int key1 = k0 + 16 + l16;
#pragma unroll
        for (int i = 0; i < 8; ++i) {
            const int qg = qrow0 + i + 8 * half;
            if (key0 > qg) s0[i] = NEG_INF_L;
            if (key1 > qg) s1[i] = NEG_INF_L;
        }

        // online softmax; row groups live in 16-lane halves (xor masks <= 8)
#pragma unroll
        for (int i = 0; i < 8; ++i) {
            float smax = fmaxf(s0[i], s1[i]);
            smax = fmaxf(smax, __shfl_xor(smax, 8, 32));
            smax = fmaxf(smax, __shfl_xor(smax, 4, 32));
            smax = fmaxf(smax, __shfl_xor(smax, 2, 32));
            smax = fmaxf(smax, __shfl_xor(smax, 1, 32));
            const float mnew = fmaxf(mrow[i], smax);
            const float p0 = __expf(s0[i] - mnew);
            const float p1 = __expf(s1[i] - mnew);
            const float corr = __expf(mrow[i] - mnew);
            float psum = p0 + p1;
            psum += __shfl_xor(psum, 8, 32);
            psum += __shfl_xor(psum, 4, 32);
            psum += __shfl_xor(psum, 2, 32);
            psum += __shfl_xor(psum, 1, 32);
            lrow[i] = lrow[i] * corr + psum;
            mrow[i] = mnew;
#pragma unroll
            for (int t = 0; t < 4; ++t) acc[t][i] *= corr;
            const int r = i + 8 * half;
            Psh[wave][r][l16]      = (__bf16)p0;   // C layout -> LDS
            Psh[wave][r][16 + l16] = (__bf16)p1;
        }

        // wave-private LDS bounce: read P back as an A fragment (DS in-order)
        v16bf pa;
#pragma unroll
        for (int v = 0; v < 4; ++v) {
            int kp = 2 * v + 8 * half;
            pa[2 * v]     = Psh[wave][l16][kp];
            pa[2 * v + 1] = Psh[wave][l16][kp + 1];
            pa[2 * v + 8] = Psh[wave][l16][16 + kp];
            pa[2 * v + 9] = Psh[wave][l16][16 + kp + 1];
        }

        // O += P V  (K-dim = 32 keys; 4 d-tiles of 16)
#pragma unroll
        for (int t = 0; t < 4; ++t) {
            v16bf vb;
#pragma unroll
            for (int j = 0; j < 8; ++j) {
                int kk = (j >> 2) * 16 + 8 * half + 2 * (j & 3);
                vb[2 * j]     = Vt[t * 16 + l16][kk];
                vb[2 * j + 1] = Vt[t * 16 + l16][kk + 1];
            }
            acc[t] = __builtin_amdgcn_wmma_f32_16x16x32_bf16(
                false, pa, false, vb, (short)0, acc[t], false, false);
        }
        __syncthreads();
    }

    // epilogue: normalize and scatter (fp32)
#pragma unroll
    for (int t = 0; t < 4; ++t) {
#pragma unroll
        for (int i = 0; i < 8; ++i) {
            const int row = qrow0 + i + 8 * half;
            const int col = head * HEAD_DIM + t * 16 + l16;
            O[(size_t)(bi * SEQ_N + row) * D_MODEL + col] = acc[t][i] / lrow[i];
        }
    }
}

// ---------------------------------------------------------------------------
// LayerNorm over last dim (1024), one block per row -> bf16 output
// ---------------------------------------------------------------------------
__global__ __launch_bounds__(256)
void layernorm_bf16(const float* __restrict__ X, const float* __restrict__ g,
                    const float* __restrict__ beta, __bf16* __restrict__ Y) {
    const int row = blockIdx.x;
    const float* __restrict__ xr = X + (size_t)row * D_MODEL;

    float s = 0.0f, ss = 0.0f;
    for (int i = threadIdx.x; i < D_MODEL; i += 256) {
        const float v = xr[i];
        s += v; ss += v * v;
    }
    // wave32 reduce
    for (int off = 16; off >= 1; off >>= 1) {
        s  += __shfl_xor(s,  off, 32);
        ss += __shfl_xor(ss, off, 32);
    }
    __shared__ float redS[8], redQ[8], stat[2];
    const int wave = threadIdx.x >> 5, lane = threadIdx.x & 31;
    if (lane == 0) { redS[wave] = s; redQ[wave] = ss; }
    __syncthreads();
    if (threadIdx.x == 0) {
        float ts = 0.0f, tq = 0.0f;
#pragma unroll
        for (int w = 0; w < 8; ++w) { ts += redS[w]; tq += redQ[w]; }
        const float mean = ts * (1.0f / D_MODEL);
        const float var  = tq * (1.0f / D_MODEL) - mean * mean;
        stat[0] = mean;
        stat[1] = rsqrtf(var + 1e-5f);
    }
    __syncthreads();
    const float mean = stat[0], inv = stat[1];
    for (int i = threadIdx.x; i < D_MODEL; i += 256)
        Y[(size_t)row * D_MODEL + i] = (__bf16)((xr[i] - mean) * inv * g[i] + beta[i]);
}

// ---------------------------------------------------------------------------
// Host launcher
// ---------------------------------------------------------------------------
extern "C" void kernel_launch(void* const* d_in, const int* in_sizes, int n_in,
                              void* d_out, int out_size, void* d_ws, size_t ws_size,
                              hipStream_t stream) {
    const float* x    = (const float*)d_in[0];
    const float* Wq   = (const float*)d_in[1];
    const float* bq   = (const float*)d_in[2];
    const float* Wk   = (const float*)d_in[3];
    const float* bk   = (const float*)d_in[4];
    const float* Wv   = (const float*)d_in[5];
    const float* bv   = (const float*)d_in[6];
    const float* ln_g = (const float*)d_in[7];
    const float* ln_b = (const float*)d_in[8];
    const float* Wo   = (const float*)d_in[9];
    const float* bo   = (const float*)d_in[10];
    float* out = (float*)d_out;

    // workspace layout (bytes)
    char* ws = (char*)d_ws;
    size_t off = 0;
    __bf16* xb   = (__bf16*)(ws + off); off += (size_t)ROWS * D_MODEL * 2;   // 8 MB
    __bf16* WqT  = (__bf16*)(ws + off); off += (size_t)D_MODEL * D_MODEL * 2;
    __bf16* WkT  = (__bf16*)(ws + off); off += (size_t)KV_DIM * D_MODEL * 2;
    __bf16* WvT  = (__bf16*)(ws + off); off += (size_t)KV_DIM * D_MODEL * 2;
    __bf16* WoT  = (__bf16*)(ws + off); off += (size_t)D_MODEL * D_MODEL * 2;
    __bf16* Qbuf = (__bf16*)(ws + off); off += (size_t)ROWS * D_MODEL * 2;
    __bf16* Kbuf = (__bf16*)(ws + off); off += (size_t)ROWS * KV_DIM * 2;
    __bf16* Vbuf = (__bf16*)(ws + off); off += (size_t)ROWS * KV_DIM * 2;
    float*  Abuf = (float*) (ws + off); off += (size_t)ROWS * D_MODEL * 4;   // attn out
    __bf16* Lbuf = (__bf16*)(ws + off); off += (size_t)ROWS * D_MODEL * 2;   // LN out
    (void)ws_size; (void)n_in; (void)in_sizes; (void)out_size;

    // ---- prep: bf16 conversions / weight transposes
    cvt_f32_to_bf16<<<1024, 256, 0, stream>>>(x, xb, ROWS * D_MODEL);
    transpose_w_bf16<<<1024, 256, 0, stream>>>(Wq, WqT, D_MODEL, D_MODEL);
    transpose_w_bf16<<<512,  256, 0, stream>>>(Wk, WkT, D_MODEL, KV_DIM);
    transpose_w_bf16<<<512,  256, 0, stream>>>(Wv, WvT, D_MODEL, KV_DIM);
    transpose_w_bf16<<<1024, 256, 0, stream>>>(Wo, WoT, D_MODEL, D_MODEL);

    // ---- QKV projections (Q pre-scaled by 1/sqrt(head_dim))
    const float qscale = 0.125f;  // 1/sqrt(64)
    gemm_bf16_wmma<<<dim3(ROWS / 256, D_MODEL / 64), 256, 0, stream>>>(
        xb, WqT, bq, (void*)Qbuf, ROWS, D_MODEL, D_MODEL, /*bf16*/1, qscale);
    gemm_bf16_wmma<<<dim3(ROWS / 256, KV_DIM / 64), 256, 0, stream>>>(
        xb, WkT, bk, (void*)Kbuf, ROWS, KV_DIM, D_MODEL, 1, 1.0f);
    gemm_bf16_wmma<<<dim3(ROWS / 256, KV_DIM / 64), 256, 0, stream>>>(
        xb, WvT, bv, (void*)Vbuf, ROWS, KV_DIM, D_MODEL, 1, 1.0f);

    // ---- causal flash attention
    attn_flash_wmma<<<dim3(SEQ_N / 64, Q_HEADS, BATCH), 128, 0, stream>>>(
        Qbuf, Kbuf, Vbuf, Abuf);

    // ---- LayerNorm -> bf16
    layernorm_bf16<<<ROWS, 256, 0, stream>>>(Abuf, ln_g, ln_b, Lbuf);

    // ---- output projection -> fp32 d_out
    gemm_bf16_wmma<<<dim3(ROWS / 256, D_MODEL / 64), 256, 0, stream>>>(
        Lbuf, WoT, bo, (void*)out, ROWS, D_MODEL, D_MODEL, /*fp32*/0, 1.0f);
}
